// CompressionTransformer_28149215658108
// MI455X (gfx1250) — matrix operand order repaired and checked by reading, not compiled, not run on hardware
//
#include <hip/hip_runtime.h>
#include <hip/hip_bf16.h>
#include <math.h>
#include <stdint.h>

typedef __attribute__((ext_vector_type(16))) __bf16 v16bf;
typedef __attribute__((ext_vector_type(8)))  __bf16 v8bf;
typedef __attribute__((ext_vector_type(8)))  float  v8f;
typedef __attribute__((ext_vector_type(4)))  unsigned int u32x4;
typedef __attribute__((ext_vector_type(4)))  int i32x4;
typedef __attribute__((ext_vector_type(8)))  int i32x8;

#define BB   2
#define SS   1024
#define DDIM 512
#define HHD  8
#define DH_  64
#define CC   64
#define MLPD 1024
#define OUTD 32000
#define EPSF 1e-5f
#define NROW (BB*SS)   // 2048

// ---------------------------------------------------------------------------
// TDM probe: plausible 2D D# per ISA §8.3/8.4 (compiled into module; not launched).
// Loads a 128-row x 32-elem bf16 tile (tensor 1024x512, row stride 512) to LDS.
// ---------------------------------------------------------------------------
__global__ void tdm_probe_kernel(const __bf16* __restrict__ src, float* __restrict__ dst) {
  __shared__ __align__(16) __bf16 tile[128 * 32];
  if (threadIdx.x == 0) {
    unsigned lds = (unsigned)(uintptr_t)&tile[0];
    unsigned long long ga = (unsigned long long)(uintptr_t)src;
    u32x4 g0;
    g0[0] = 1u;                                     // count=1 user descriptor
    g0[1] = lds;                                    // lds_addr [63:32]
    g0[2] = (unsigned)(ga & 0xffffffffu);           // global_addr [95:64]
    g0[3] = (unsigned)((ga >> 32) & 0x01ffffffu)    // global_addr [120:96]
          | (2u << 30);                             // type=2 ("image") [127:126]
    i32x8 g1;
    g1[0] = (1 << 16);                              // wg_mask=0, data_size=1 (2B)
    g1[1] = (512 & 0xffff) << 16;                   // tensor_dim0 lo16 @ [63:48]
    g1[2] = (512 >> 16) | ((1024 & 0xffff) << 16);  // tensor_dim0 hi16 | tensor_dim1 lo16
    g1[3] = (1024 >> 16) | (32 << 16);              // tensor_dim1 hi16 | tile_dim0=32
    g1[4] = 128;                                    // tile_dim1=128, tile_dim2=0
    g1[5] = 512;                                    // tensor_dim0_stride lo32 (=K)
    g1[6] = 0;                                      // stride0 hi16 | stride1 lo16
    g1[7] = 0;                                      // stride1 hi32
    i32x4 g2 = {};
    i32x4 g3 = {};
#if defined(__clang_major__) && (__clang_major__ >= 23)
    i32x8 g4 = {};
    __builtin_amdgcn_tensor_load_to_lds(g0, g1, g2, g3, g4, 0);
#else
    __builtin_amdgcn_tensor_load_to_lds(g0, g1, g2, g3, 0);
#endif
    __builtin_amdgcn_s_wait_tensorcnt(0);
  }
  __syncthreads();
  dst[threadIdx.x] = (float)tile[threadIdx.x];
}

// ---------------------------------------------------------------------------
// Weight convert + transpose: W (rows x cols, f32 row-major) -> WT (cols x rows, bf16)
// ---------------------------------------------------------------------------
__global__ void wT_bf16_kernel(const float* __restrict__ W, __bf16* __restrict__ WT,
                               int rows, int cols) {
  size_t idx = (size_t)blockIdx.x * blockDim.x + threadIdx.x;
  size_t total = (size_t)rows * cols;
  if (idx >= total) return;
  int n = (int)(idx / rows);
  int k = (int)(idx % rows);
  WT[idx] = (__bf16)W[(size_t)k * cols + n];
}

// ---------------------------------------------------------------------------
// Embedding gather + LayerNorm -> f32 + bf16 (one 256-thr block per row, D=512)
// ---------------------------------------------------------------------------
__global__ void embed_ln_kernel(const int* __restrict__ tokens, const float* __restrict__ emb,
                                const float* __restrict__ g, const float* __restrict__ bta,
                                float* __restrict__ outF, __bf16* __restrict__ outB) {
  __shared__ float red[256];
  int row = blockIdx.x, tid = threadIdx.x;
  const float* src = emb + (size_t)tokens[row] * DDIM;
  float v0 = src[tid], v1 = src[tid + 256];
  red[tid] = v0 + v1; __syncthreads();
  for (int s = 128; s > 0; s >>= 1) { if (tid < s) red[tid] += red[tid + s]; __syncthreads(); }
  float mean = red[0] * (1.0f / DDIM); __syncthreads();
  float d0 = v0 - mean, d1 = v1 - mean;
  red[tid] = d0 * d0 + d1 * d1; __syncthreads();
  for (int s = 128; s > 0; s >>= 1) { if (tid < s) red[tid] += red[tid + s]; __syncthreads(); }
  float inv = rsqrtf(red[0] * (1.0f / DDIM) + EPSF);
  float o0 = d0 * inv * g[tid]       + bta[tid];
  float o1 = d1 * inv * g[tid + 256] + bta[tid + 256];
  size_t off = (size_t)row * DDIM;
  outF[off + tid] = o0; outF[off + tid + 256] = o1;
  outB[off + tid] = (__bf16)o0; outB[off + tid + 256] = (__bf16)o1;
}

// ---------------------------------------------------------------------------
// (in1 [+ in2]) -> LayerNorm -> optional f32 / bf16 outputs
// ---------------------------------------------------------------------------
__global__ void addln_kernel(const float* __restrict__ in1, const float* __restrict__ in2,
                             const float* __restrict__ g, const float* __restrict__ bta,
                             float* __restrict__ outF, __bf16* __restrict__ outB) {
  __shared__ float red[256];
  int row = blockIdx.x, tid = threadIdx.x;
  size_t off = (size_t)row * DDIM;
  float v0 = in1[off + tid]       + (in2 ? in2[off + tid]       : 0.f);
  float v1 = in1[off + tid + 256] + (in2 ? in2[off + tid + 256] : 0.f);
  red[tid] = v0 + v1; __syncthreads();
  for (int s = 128; s > 0; s >>= 1) { if (tid < s) red[tid] += red[tid + s]; __syncthreads(); }
  float mean = red[0] * (1.0f / DDIM); __syncthreads();
  float d0 = v0 - mean, d1 = v1 - mean;
  red[tid] = d0 * d0 + d1 * d1; __syncthreads();
  for (int s = 128; s > 0; s >>= 1) { if (tid < s) red[tid] += red[tid + s]; __syncthreads(); }
  float inv = rsqrtf(red[0] * (1.0f / DDIM) + EPSF);
  float o0 = d0 * inv * g[tid]       + bta[tid];
  float o1 = d1 * inv * g[tid + 256] + bta[tid + 256];
  if (outF) { outF[off + tid] = o0; outF[off + tid + 256] = o1; }
  if (outB) { outB[off + tid] = (__bf16)o0; outB[off + tid + 256] = (__bf16)o1; }
}

// ---------------------------------------------------------------------------
// Build a v16bf WMMA fragment from two 16B LDS chunks.
// ---------------------------------------------------------------------------
__device__ __forceinline__ v16bf make_frag(const __bf16* p0, const __bf16* p1) {
  v8bf lo = *(const v8bf*)p0;
  v8bf hi = *(const v8bf*)p1;
  v16bf r;
#pragma unroll
  for (int i = 0; i < 8; i++) { r[i] = lo[i]; r[i + 8] = hi[i]; }
  return r;
}

// ---------------------------------------------------------------------------
// WMMA bf16 GEMM with CDNA5 async global->LDS double buffering.
// out = A(MxK bf16 row-major) * BT(NxK bf16 row-major)^T + bias
// 256 thr = 8 waves as 2(M) x 4(N); wave tile 32x32 (4 accumulators);
// block tile 64x128; K step 32.
// Staging: per k-tile each thread issues 3 global_load_async_to_lds_b128
// (1 A chunk + 2 B chunks); s_wait_asynccnt 3 overlaps tile t+1 with tile t.
// LDS rows padded to 40 bf16 (80B) to spread b128 reads across the 64 banks.
// Fragment layouts per §7.12.2:
//   A (16x32): lane l/l+16 -> row; K chunks {hf*8..+8} and {16+hf*8..+8}
//   B (32x16): lane -> col;  K chunks {hf*8..+8} and {16+hf*8..+8}
//   C/D (16x16 f32): VGPR j -> (m + hf*8 + j, n + l)
// ---------------------------------------------------------------------------
__global__ void gemm_bf16_kernel(const __bf16* __restrict__ A, const __bf16* __restrict__ BT,
                                 const float* __restrict__ bias,
                                 float* __restrict__ outF, __bf16* __restrict__ outB,
                                 int M, int N, int K, int relu) {
  __shared__ __align__(16) __bf16 ldsA[2][64 * 40];
  __shared__ __align__(16) __bf16 ldsB[2][128 * 40];

  int tid = threadIdx.x;
  int wave = tid >> 5, lane = tid & 31;
  int wm = wave >> 2, wn = wave & 3;
  int l = lane & 15, hf = lane >> 4;
  int mblk = blockIdx.x * 64, nblk = blockIdx.y * 128;

  // staging coordinates: row 0..63, 16B chunk 0..3 (32 k-elems per row)
  int sr = tid >> 2, sc = tid & 3;
  const __bf16* gA  = A  + (size_t)(mblk + sr) * K + sc * 8;
  const __bf16* gB0 = BT + (size_t)(nblk + sr) * K + sc * 8;
  const __bf16* gB1 = BT + (size_t)(nblk + 64 + sr) * K + sc * 8;
  int laOff  = sr * 40 + sc * 8;
  int lb1Off = (sr + 64) * 40 + sc * 8;

  auto stage = [&](int buf, int k0) {
    unsigned la  = (unsigned)(uintptr_t)&ldsA[buf][laOff];
    unsigned lb0 = (unsigned)(uintptr_t)&ldsB[buf][laOff];
    unsigned lb1 = (unsigned)(uintptr_t)&ldsB[buf][lb1Off];
    const __bf16* pa  = gA  + k0;
    const __bf16* pb0 = gB0 + k0;
    const __bf16* pb1 = gB1 + k0;
    asm volatile("global_load_async_to_lds_b128 %0, %1, off" :: "v"(la),  "v"(pa)  : "memory");
    asm volatile("global_load_async_to_lds_b128 %0, %1, off" :: "v"(lb0), "v"(pb0) : "memory");
    asm volatile("global_load_async_to_lds_b128 %0, %1, off" :: "v"(lb1), "v"(pb1) : "memory");
    if (k0 + 64 < K) {                 // gfx1250 global_prefetch of tile t+2
      __builtin_prefetch(pb0 + 64, 0, 1);
      __builtin_prefetch(pb1 + 64, 0, 1);
    }
  };

  v8f acc00 = {}, acc01 = {}, acc10 = {}, acc11 = {};
  int ar0 = wm * 32 + l, ar1 = ar0 + 16;
  int br0 = wn * 32 + l, br1 = br0 + 16;

  stage(0, 0);
  int nk = K >> 5;
  for (int t = 0; t < nk; t++) {
    int buf = t & 1;
    if (t + 1 < nk) {
      stage(buf ^ 1, (t + 1) << 5);
      asm volatile("s_wait_asynccnt 0x3" ::: "memory");   // tile t landed
    } else {
      asm volatile("s_wait_asynccnt 0x0" ::: "memory");
    }
    __syncthreads();
    const __bf16* pA = &ldsA[buf][0];
    const __bf16* pB = &ldsB[buf][0];
    v16bf av0 = make_frag(pA + ar0 * 40 + hf * 8,  pA + ar0 * 40 + 16 + hf * 8);
    v16bf av1 = make_frag(pA + ar1 * 40 + hf * 8,  pA + ar1 * 40 + 16 + hf * 8);
    v16bf bv0 = make_frag(pB + br0 * 40 + hf * 8,  pB + br0 * 40 + 16 + hf * 8);
    v16bf bv1 = make_frag(pB + br1 * 40 + hf * 8,  pB + br1 * 40 + 16 + hf * 8);
    acc00 = __builtin_amdgcn_wmma_f32_16x16x32_bf16(false, av0, false, bv0, (short)0, acc00, false, false);
    acc01 = __builtin_amdgcn_wmma_f32_16x16x32_bf16(false, av0, false, bv1, (short)0, acc01, false, false);
    acc10 = __builtin_amdgcn_wmma_f32_16x16x32_bf16(false, av1, false, bv0, (short)0, acc10, false, false);
    acc11 = __builtin_amdgcn_wmma_f32_16x16x32_bf16(false, av1, false, bv1, (short)0, acc11, false, false);
    __syncthreads();   // all waves done reading buf before it is restaged
  }

  int m0 = mblk + wm * 32;
  int n0 = nblk + wn * 32;
  float bias0 = bias ? bias[n0 + l]      : 0.0f;
  float bias1 = bias ? bias[n0 + 16 + l] : 0.0f;
#pragma unroll
  for (int j = 0; j < 8; j++) {
    int r0 = m0 + hf * 8 + j;
    int r1 = m0 + 16 + hf * 8 + j;
    float v00 = acc00[j] + bias0, v01 = acc01[j] + bias1;
    float v10 = acc10[j] + bias0, v11 = acc11[j] + bias1;
    if (relu) {
      v00 = fmaxf(v00, 0.f); v01 = fmaxf(v01, 0.f);
      v10 = fmaxf(v10, 0.f); v11 = fmaxf(v11, 0.f);
    }
    size_t i00 = (size_t)r0 * N + (n0 + l);
    size_t i01 = (size_t)r0 * N + (n0 + 16 + l);
    size_t i10 = (size_t)r1 * N + (n0 + l);
    size_t i11 = (size_t)r1 * N + (n0 + 16 + l);
    if (outF) { outF[i00] = v00; outF[i01] = v01; outF[i10] = v10; outF[i11] = v11; }
    if (outB) {
      outB[i00] = (__bf16)v00; outB[i01] = (__bf16)v01;
      outB[i10] = (__bf16)v10; outB[i11] = (__bf16)v11;
    }
  }
}

// ---------------------------------------------------------------------------
// RoPE in-place on kd (f32, (B*S, 512)); first ROT=32 dims of each 64-dim head.
// ---------------------------------------------------------------------------
__global__ void rope_kernel(float* __restrict__ kd) {
  int idx = blockIdx.x * blockDim.x + threadIdx.x;   // B*S*H*16 = 262144
  int j = idx & 15;
  int h = (idx >> 4) & 7;
  int row = idx >> 7;                                // 0..2047
  int s = row & (SS - 1);
  float ang = (float)s * __powf(10000.0f, -(2.0f * (float)j) / 32.0f);
  float c = __cosf(ang), sn = __sinf(ang);
  float* p = kd + (size_t)row * DDIM + h * DH_ + 2 * j;
  float x0 = p[0], x1 = p[1];
  p[0] = x0 * c - x1 * sn;
  p[1] = x1 * c + x0 * sn;
}

// ---------------------------------------------------------------------------
// down_t[(bh*S + s)*64 + c] = scale * dot_d( q_down[c, h*64+d], kd[b,s,h*64+d] )
// ---------------------------------------------------------------------------
__global__ void down_kernel(const float* __restrict__ q_down, const float* __restrict__ kd,
                            float* __restrict__ downt) {
  __shared__ float krow[DH_];
  int c = threadIdx.x;
  int gid = blockIdx.x;
  int s  = gid & (SS - 1);
  int bh = gid >> 10;          // b*H + h
  int b = bh >> 3, h = bh & 7;
  const float* kp = kd + ((size_t)(b * SS + s)) * DDIM + h * DH_;
  krow[c] = kp[c];
  __syncthreads();
  const float* qp = q_down + (size_t)c * DDIM + h * DH_;
  float acc = 0.f;
#pragma unroll 8
  for (int d = 0; d < DH_; d++) acc += qp[d] * krow[d];
  downt[((size_t)bh * SS + s) * CC + c] = acc * 0.125f;   // DH^-0.5
}

// ---------------------------------------------------------------------------
// Sequential compression-attention scan: one block (256 thr) per (b,h).
// ---------------------------------------------------------------------------
__global__ void scan_kernel(const float* __restrict__ downt, const float* __restrict__ qu,
                            const float* __restrict__ vdk, const float* __restrict__ vdv,
                            __bf16* __restrict__ attnb) {
  __shared__ float acc[CC][2 * DH_];
  __shared__ float maxc[CC], normc[CC], wsh[CC], qus[DH_], vks[2 * DH_], upc[CC], red[256];
  __shared__ float invs;

  int bh = blockIdx.x;
  int b = bh >> 3, h = bh & 7;
  int tid = threadIdx.x;
  const float* dt = downt + (size_t)bh * SS * CC;

  if (tid < CC) {
    float m = -1e30f;
    for (int s = 0; s < SS; s++) m = fmaxf(m, dt[s * CC + tid]);
    maxc[tid] = m;
    normc[tid] = 0.f;
  }
  for (int i = tid; i < CC * 2 * DH_; i += 256) acc[i >> 7][i & 127] = 0.f;
  __syncthreads();

  for (int s = 0; s < SS; s++) {
    size_t rowoff = ((size_t)(b * SS + s)) * DDIM + h * DH_;
    if (tid < 64) {
      wsh[tid] = __expf(dt[s * CC + tid] - maxc[tid]);
      qus[tid] = qu[rowoff + tid] * 0.125f;
    } else if (tid < 128) {
      vks[tid - 64] = vdk[rowoff + (tid - 64)];
    } else if (tid < 192) {
      vks[DH_ + (tid - 128)] = vdv[rowoff + (tid - 128)];
    }
    __syncthreads();
    {
      int c = tid >> 2, q = tid & 3;
      float w = wsh[c];
#pragma unroll
      for (int d = q * 32; d < q * 32 + 32; d++) acc[c][d] += w * vks[d];
    }
    if (tid < CC) normc[tid] += wsh[tid];
    __syncthreads();
    {
      int c = tid >> 2, q = tid & 3;
      float part = 0.f;
#pragma unroll
      for (int d = q * 16; d < q * 16 + 16; d++) part += qus[d] * acc[c][d];
      red[tid] = part;
    }
    __syncthreads();
    if (tid < CC)
      upc[tid] = (red[4 * tid] + red[4 * tid + 1] + red[4 * tid + 2] + red[4 * tid + 3]) / normc[tid];
    __syncthreads();
    if (tid == 0) {
      float m = upc[0];
      for (int i = 1; i < CC; i++) m = fmaxf(m, upc[i]);
      float ssum = 0.f;
      for (int i = 0; i < CC; i++) { float e = __expf(upc[i] - m); upc[i] = e; ssum += e; }
      invs = 1.0f / ssum;
    }
    __syncthreads();
    {
      int d = tid >> 2, q = tid & 3;
      float part = 0.f;
#pragma unroll
      for (int c = q * 16; c < q * 16 + 16; c++) part += upc[c] * acc[c][DH_ + d] / normc[c];
      red[tid] = part;
    }
    __syncthreads();
    if (tid < DH_) {
      float o = (red[4 * tid] + red[4 * tid + 1] + red[4 * tid + 2] + red[4 * tid + 3]) * invs;
      attnb[rowoff + tid] = (__bf16)o;
    }
    __syncthreads();
  }
}

// ---------------------------------------------------------------------------
extern "C" void kernel_launch(void* const* d_in, const int* in_sizes, int n_in,
                              void* d_out, int out_size, void* d_ws, size_t ws_size,
                              hipStream_t stream) {
  const int*   tokens = (const int*)  d_in[0];
  const float* emb    = (const float*)d_in[1];
  const float* ln1_g  = (const float*)d_in[2];
  const float* ln1_b  = (const float*)d_in[3];
  const float* q_down = (const float*)d_in[4];
  const float* Wq     = (const float*)d_in[5];
  const float* bq     = (const float*)d_in[6];
  const float* Wk     = (const float*)d_in[7];
  const float* bk     = (const float*)d_in[8];
  const float* Wv     = (const float*)d_in[9];
  const float* bv     = (const float*)d_in[10];
  const float* Wo     = (const float*)d_in[11];
  const float* bo     = (const float*)d_in[12];
  const float* ln2_g  = (const float*)d_in[13];
  const float* ln2_b  = (const float*)d_in[14];
  const float* W1     = (const float*)d_in[15];
  const float* b1     = (const float*)d_in[16];
  const float* W2     = (const float*)d_in[17];
  const float* b2     = (const float*)d_in[18];
  const float* lnf_g  = (const float*)d_in[19];
  const float* lnf_b  = (const float*)d_in[20];
  const float* Wout   = (const float*)d_in[21];

  char* ws = (char*)d_ws;
  size_t off = 0;
  auto take = [&](size_t bytes) { char* p = ws + off; off = (off + bytes + 255) & ~(size_t)255; return p; };

  __bf16* WqT   = (__bf16*)take((size_t)DDIM * DDIM * 2);
  __bf16* WkT   = (__bf16*)take((size_t)DDIM * DDIM * 2);
  __bf16* WvT   = (__bf16*)take((size_t)DDIM * DDIM * 2);
  __bf16* WoT   = (__bf16*)take((size_t)DDIM * DDIM * 2);
  __bf16* W1T   = (__bf16*)take((size_t)MLPD * DDIM * 2);
  __bf16* W2T   = (__bf16*)take((size_t)DDIM * MLPD * 2);
  __bf16* WouT  = (__bf16*)take((size_t)OUTD * DDIM * 2);
  float*  hF    = (float*) take((size_t)NROW * DDIM * 4);
  __bf16* hB    = (__bf16*)take((size_t)NROW * DDIM * 2);
  float*  quF   = (float*) take((size_t)NROW * DDIM * 4);
  float*  kdF   = (float*) take((size_t)NROW * DDIM * 4);
  __bf16* vdB   = (__bf16*)take((size_t)NROW * DDIM * 2);
  float*  vdkF  = (float*) take((size_t)NROW * DDIM * 4);
  float*  vdvF  = (float*) take((size_t)NROW * DDIM * 4);
  float*  downt = (float*) take((size_t)BB * HHD * SS * CC * 4);
  __bf16* attnB = (__bf16*)take((size_t)NROW * DDIM * 2);
  float*  aF    = (float*) take((size_t)NROW * DDIM * 4);
  float*  xF    = (float*) take((size_t)NROW * DDIM * 4);
  __bf16* xB    = (__bf16*)take((size_t)NROW * DDIM * 2);
  __bf16* m1B   = (__bf16*)take((size_t)NROW * MLPD * 2);
  float*  ffF   = (float*) take((size_t)NROW * DDIM * 4);
  __bf16* xfB   = (__bf16*)take((size_t)NROW * DDIM * 2);

  auto T = [&](const float* W, __bf16* WT, int rows, int cols) {
    size_t total = (size_t)rows * cols;
    int blocks = (int)((total + 255) / 256);
    wT_bf16_kernel<<<blocks, 256, 0, stream>>>(W, WT, rows, cols);
  };
  auto G = [&](const __bf16* A, const __bf16* BT, const float* bias,
               float* oF, __bf16* oB, int M, int N, int K, int relu) {
    dim3 grid(M / 64, N / 128);
    gemm_bf16_kernel<<<grid, 256, 0, stream>>>(A, BT, bias, oF, oB, M, N, K, relu);
  };

  // Weight conversion (bf16, transposed to N-major for contiguous WMMA B fragments)
  T(Wq,   WqT,  DDIM, DDIM);
  T(Wk,   WkT,  DDIM, DDIM);
  T(Wv,   WvT,  DDIM, DDIM);
  T(Wo,   WoT,  DDIM, DDIM);
  T(W1,   W1T,  DDIM, MLPD);
  T(W2,   W2T,  MLPD, DDIM);
  T(Wout, WouT, DDIM, OUTD);

  // h = LN(emb[tokens])
  embed_ln_kernel<<<NROW, 256, 0, stream>>>(tokens, emb, ln1_g, ln1_b, hF, hB);

  // Projections
  G(hB, WqT, bq, quF,  nullptr, NROW, DDIM, DDIM, 0);   // qu (f32 for scan)
  G(hB, WkT, bk, kdF,  nullptr, NROW, DDIM, DDIM, 0);   // kd (f32, RoPE'd next)
  G(hB, WvT, bv, nullptr, vdB,  NROW, DDIM, DDIM, 0);   // vd (bf16, GEMM input)
  G(vdB, WkT, bk, vdkF, nullptr, NROW, DDIM, DDIM, 0);  // vdk
  G(vdB, WvT, bv, vdvF, nullptr, NROW, DDIM, DDIM, 0);  // vdv

  // RoPE on kd
  rope_kernel<<<(BB * SS * HHD * 16) / 256, 256, 0, stream>>>(kdF);

  // down scores + sequential compression scan
  down_kernel<<<BB * HHD * SS, 64, 0, stream>>>(q_down, kdF, downt);
  scan_kernel<<<BB * HHD, 256, 0, stream>>>(downt, quF, vdkF, vdvF, attnB);

  // output proj, residual + LN2
  G(attnB, WoT, bo, aF, nullptr, NROW, DDIM, DDIM, 0);
  addln_kernel<<<NROW, 256, 0, stream>>>(hF, aF, ln2_g, ln2_b, xF, xB);

  // MLP
  G(xB,  W1T, b1, nullptr, m1B, NROW, MLPD, DDIM, 1);   // relu(x@W1+b1)
  G(m1B, W2T, b2, ffF, nullptr, NROW, DDIM, MLPD, 0);   // @W2+b2

  // residual + final LN
  addln_kernel<<<NROW, 256, 0, stream>>>(xF, ffF, lnf_g, lnf_b, nullptr, xfB);

  // logits
  G(xfB, WouT, nullptr, (float*)d_out, nullptr, NROW, OUTD, DDIM, 0);
}
